// CustomLSTM_67001489818092
// MI455X (gfx1250) — compile-verified, matrix operand order checked by reading, not run on hardware
//
#include <hip/hip_runtime.h>
#include <hip/hip_bf16.h>

// ---------------------------------------------------------------------------
// 2-layer LSTM for MI455X (gfx1250), bf16 WMMA (v_wmma_f32_16x16x32_bf16)
// T=512, B=128, H=1024, 4H=4096, L=2
//
// Phase 1: pack Wih/Whh f32 -> bf16 in WMMA B-fragment layout.
// Phase 2: big WMMA GEMM  gx = x @ Wih + bias   (M=65536, K=1024, N=4096)
// Phase 3: 512 sequential step kernels/layer: gates = gx(t) + h @ Whh (WMMA,
//          C seeded from gx), fused LSTM cell + length masking.
//
// B tiles are double-buffered in LDS via GLOBAL_LOAD_ASYNC_TO_LDS_B128
// (ASYNCcnt), removing the 8x per-block B-load redundancy from L2.
// ---------------------------------------------------------------------------

typedef __attribute__((ext_vector_type(16))) __bf16 v16bf;
typedef __attribute__((ext_vector_type(8)))  float  v8f;

#define T_STEPS 512
#define BATCH   128
#define HID     1024
#define GATES   4096
#define NLAYER  2

// K offset inside a 16x32 bf16 A/B fragment for (lane, element i).
__device__ __forceinline__ int koff_lane(int lane, int i) {
  int kh = lane >> 4;          // 0 or 1
  int v = i >> 1, sub = i & 1; // 8 VGPRs x 2 bf16
  return (v < 4) ? (kh * 8 + v * 2 + sub) : (16 + kh * 8 + (v - 4) * 2 + sub);
}

__device__ __forceinline__ float sigmoidf_(float x) {
  return 1.0f / (1.0f + expf(-x));
}

// Async copy of 16 bytes: LDS[lds_byte_off] <= global[g]
__device__ __forceinline__ void async_ld_b128(unsigned lds_byte_off,
                                              const char* g) {
  asm volatile("global_load_async_to_lds_b128 %0, %1, off"
               :: "v"(lds_byte_off), "v"(g)
               : "memory");
}
__device__ __forceinline__ void wait_async0() {
  asm volatile("s_wait_asynccnt 0x0" ::: "memory");
}

// --- pack a [K,N] f32 matrix into bf16 WMMA B fragments -------------------
// packed index = (((fk * (N/16) + fn) * 32) + lane) * 16 + i
__global__ void pack_weight_kernel(const float* __restrict__ W,
                                   __bf16* __restrict__ P, int K, int N) {
  size_t idx = (size_t)blockIdx.x * 256 + threadIdx.x;
  size_t total = (size_t)K * N;
  if (idx >= total) return;
  int i    = (int)(idx & 15);
  int lane = (int)((idx >> 4) & 31);
  size_t rest = idx >> 9;
  int NF = N / 16;
  int fn = (int)(rest % NF);
  int fk = (int)(rest / NF);
  int n = fn * 16 + (lane & 15);
  int k = fk * 32 + koff_lane(lane, i);
  P[idx] = (__bf16)W[(size_t)k * N + n];
}

__global__ void f32_to_bf16_kernel(const float* __restrict__ src,
                                   __bf16* __restrict__ dst, size_t n) {
  size_t idx = (size_t)blockIdx.x * 256 + threadIdx.x;
  if (idx < n) dst[idx] = (__bf16)src[idx];
}

__global__ void zero_states_kernel(__bf16* __restrict__ hbf,
                                   float* __restrict__ hf,
                                   float* __restrict__ cf) {
  int idx = blockIdx.x * 256 + threadIdx.x; // BATCH*HID threads
  hbf[idx] = (__bf16)0.0f;
  hf[idx] = 0.0f;
  cf[idx] = 0.0f;
}

__global__ void copy_states_kernel(const float* __restrict__ hf,
                                   const float* __restrict__ cf,
                                   float* __restrict__ h_out,
                                   float* __restrict__ c_out) {
  int idx = blockIdx.x * 256 + threadIdx.x; // BATCH*HID threads
  h_out[idx] = hf[idx];
  c_out[idx] = cf[idx];
}

// --- gx = A @ Wih + bias,  A:[M,1024] bf16, out bf16 [M,4096] -------------
// grid: (GATES/128, M/128), block 256 (8 waves). Wave w: rows w*16..+16,
// 8 N-fragments of 16 cols => 128x128 tile. B tile (8KB/k-iter) staged in
// LDS (double-buffered, async).
__global__ __launch_bounds__(256)
void gx_gemm_kernel(const __bf16* __restrict__ Abf,
                    const __bf16* __restrict__ BP,      // packed [1024,4096]
                    const float*  __restrict__ bias,    // [4096]
                    __bf16* __restrict__ gx) {          // [M,4096]
  __shared__ __align__(32) __bf16 smem[2 * 4096];       // 2 x 8KB

  const int tid  = threadIdx.x;
  const int lane = tid & 31;
  const int w    = tid >> 5;
  const int tileN = blockIdx.x * 128;
  const size_t rowBase = (size_t)blockIdx.y * 128 + (size_t)w * 16;
  const int mlo = lane & 15;
  const int mhiOff = (lane >> 4) * 8;
  const __bf16* arow = Abf + (rowBase + mlo) * HID;
  const int NF = GATES / 16;
  const char* BPc = (const char*)BP;

  // per-thread async-copy assignment: two 16B chunks (groups gA and gA+2)
  const int gA = tid >> 7;            // 0..1
  const unsigned coff = (unsigned)(tid & 127) * 16;
  const int fnBaseA = blockIdx.x * 8 + gA * 2;        // 2 frags per group
  const int fnBaseB = blockIdx.x * 8 + (gA + 2) * 2;

  auto stage = [&](int fk, int buf) {
    unsigned lbase = (unsigned)(uintptr_t)&smem[buf * 4096];
    size_t ga = ((size_t)fk * NF + (size_t)fnBaseA) * 1024 + coff;
    size_t gb = ((size_t)fk * NF + (size_t)fnBaseB) * 1024 + coff;
    async_ld_b128(lbase + (unsigned)gA * 2048 + coff, BPc + ga);
    async_ld_b128(lbase + (unsigned)(gA + 2) * 2048 + coff, BPc + gb);
  };

  v8f acc[8];
#pragma unroll
  for (int f = 0; f < 8; ++f)
#pragma unroll
    for (int e = 0; e < 8; ++e) acc[f][e] = 0.0f;

  stage(0, 0);                        // prologue: tile 0 -> buf 0

#pragma unroll 1
  for (int it = 0; it < HID / 32; ++it) {
    const int cur = it & 1;
    wait_async0();                    // this wave's async copies landed
    __syncthreads();                  // all waves' copies landed; prev reads done
    if (it + 1 < HID / 32) stage(it + 1, cur ^ 1);

    const int kb = it * 32;
    v16bf a;
#pragma unroll
    for (int i = 0; i < 16; ++i) a[i] = arow[kb + koff_lane(lane, i)];

    const v16bf* lB = (const v16bf*)&smem[cur * 4096];
#pragma unroll
    for (int f = 0; f < 8; ++f) {
      v16bf b = lB[f * 32 + lane];    // frag f: byte f*1024 + lane*32
      acc[f] = __builtin_amdgcn_wmma_f32_16x16x32_bf16(
          false, a, false, b, (short)0, acc[f], false, false);
    }
  }

#pragma unroll
  for (int f = 0; f < 8; ++f) {
    int n = tileN + f * 16 + mlo;
    float bv = bias[n];
#pragma unroll
    for (int r = 0; r < 8; ++r) {
      size_t row = rowBase + r + mhiOff;
      gx[row * GATES + n] = (__bf16)(acc[f][r] + bv);
    }
  }
}

// --- one recurrent step: gates = gx(t) + h @ Whh; LSTM cell + masking -----
// grid 32 blocks: block j owns cols [j*32, j*32+32) of each gate group, so
// i/f/o/g for a hidden unit share lane/VGPR slots across accumulators.
__global__ __launch_bounds__(256)
void lstm_step_kernel(const __bf16* __restrict__ gx_t,    // [B,4096], bias incl.
                      const __bf16* __restrict__ WhhP,    // packed [1024,4096]
                      const __bf16* __restrict__ hbf_in,  // [B,1024]
                      const float*  __restrict__ hf_in,   // [B,1024]
                      const float*  __restrict__ cf_in,   // [B,1024]
                      const long long* __restrict__ lengths,
                      int t,
                      __bf16* __restrict__ hbf_out,
                      float*  __restrict__ hf_out,
                      float*  __restrict__ cf_out,
                      float*  __restrict__ out_f32,       // [B,H] slice or null
                      __bf16* __restrict__ out_bf) {      // [B,H] slice or null
  __shared__ __align__(32) __bf16 smem[2 * 4096];         // 2 x 8KB

  const int tid  = threadIdx.x;
  const int lane = tid & 31;
  const int w    = tid >> 5;
  const int j    = blockIdx.x;         // 0..31
  const int m0   = w * 16;
  const int mlo  = lane & 15;
  const int mhiOff = (lane >> 4) * 8;
  const __bf16* arow = hbf_in + (size_t)(m0 + mlo) * HID;
  const int NF = GATES / 16;
  const char* WPc = (const char*)WhhP;

  // group g covers gate-group g's 2 fragments: fn = g*64 + j*2 (+0/+1)
  const int gA = tid >> 7;            // 0..1
  const unsigned coff = (unsigned)(tid & 127) * 16;
  const int fnBaseA = gA * 64 + j * 2;
  const int fnBaseB = (gA + 2) * 64 + j * 2;

  auto stage = [&](int fk, int buf) {
    unsigned lbase = (unsigned)(uintptr_t)&smem[buf * 4096];
    size_t ga = ((size_t)fk * NF + (size_t)fnBaseA) * 1024 + coff;
    size_t gb = ((size_t)fk * NF + (size_t)fnBaseB) * 1024 + coff;
    async_ld_b128(lbase + (unsigned)gA * 2048 + coff, WPc + ga);
    async_ld_b128(lbase + (unsigned)(gA + 2) * 2048 + coff, WPc + gb);
  };

  // seed accumulators with gx(t)
  v8f acc[8];
#pragma unroll
  for (int f = 0; f < 8; ++f) {
    int n = (f >> 1) * HID + j * 32 + (f & 1) * 16 + mlo;
#pragma unroll
    for (int r = 0; r < 8; ++r) {
      int m = m0 + r + mhiOff;
      acc[f][r] = (float)gx_t[(size_t)m * GATES + n];
    }
  }

  stage(0, 0);                        // prologue: tile 0 -> buf 0

#pragma unroll 1
  for (int it = 0; it < HID / 32; ++it) {
    const int cur = it & 1;
    wait_async0();
    __syncthreads();
    if (it + 1 < HID / 32) stage(it + 1, cur ^ 1);

    const int kb = it * 32;
    v16bf a;
#pragma unroll
    for (int i = 0; i < 16; ++i) a[i] = arow[kb + koff_lane(lane, i)];

    const v16bf* lB = (const v16bf*)&smem[cur * 4096];
#pragma unroll
    for (int f = 0; f < 8; ++f) {
      v16bf b = lB[f * 32 + lane];
      acc[f] = __builtin_amdgcn_wmma_f32_16x16x32_bf16(
          false, a, false, b, (short)0, acc[f], false, false);
    }
  }

  // fused LSTM cell + length masking
#pragma unroll
  for (int fi = 0; fi < 2; ++fi) {
    int col = j * 32 + fi * 16 + mlo; // hidden index
#pragma unroll
    for (int r = 0; r < 8; ++r) {
      int m = m0 + r + mhiOff;
      size_t sidx = (size_t)m * HID + col;
      float gi = acc[0 + fi][r];
      float gf = acc[2 + fi][r];
      float go = acc[4 + fi][r];
      float gg = acc[6 + fi][r];
      float cprev = cf_in[sidx];
      float hprev = hf_in[sidx];
      float c1 = sigmoidf_(gf + 1.0f) * cprev + sigmoidf_(gi) * tanhf(gg);
      float h1 = sigmoidf_(go) * tanhf(c1);
      bool msk = (long long)t < lengths[m];
      float hn = msk ? h1 : hprev;
      float cn = msk ? c1 : cprev;
      float om = msk ? h1 : 0.0f;
      cf_out[sidx] = cn;
      hf_out[sidx] = hn;
      hbf_out[sidx] = (__bf16)hn;
      if (out_f32) out_f32[sidx] = om;
      if (out_bf)  out_bf[sidx]  = (__bf16)om;
    }
  }
}

// ---------------------------------------------------------------------------
extern "C" void kernel_launch(void* const* d_in, const int* in_sizes, int n_in,
                              void* d_out, int out_size, void* d_ws, size_t ws_size,
                              hipStream_t stream) {
  const float*     x       = (const float*)d_in[0];      // [T,B,H]
  const long long* lengths = (const long long*)d_in[1];  // [B] int64
  const float*     Wih     = (const float*)d_in[2];      // [L,H,4H]
  const float*     Whh     = (const float*)d_in[3];      // [L,H,4H]
  const float*     bias    = (const float*)d_in[4];      // [L,4H]
  float* out = (float*)d_out;

  const size_t WSZ  = (size_t)HID * GATES;           // 4M elems per matrix
  const size_t XSZ  = (size_t)T_STEPS * BATCH * HID; // 67M elems
  const size_t GXSZ = (size_t)T_STEPS * BATCH * GATES;
  const size_t SSZ  = (size_t)BATCH * HID;           // 128K elems

  // workspace carve-up (256B aligned)
  char* ws = (char*)d_ws;
  size_t off = 0;
  auto carve = [&](size_t bytes) -> void* {
    void* p = ws + off;
    off = (off + bytes + 255) & ~(size_t)255;
    return p;
  };
  __bf16* WihP  = (__bf16*)carve(NLAYER * WSZ * sizeof(__bf16));
  __bf16* WhhP  = (__bf16*)carve(NLAYER * WSZ * sizeof(__bf16));
  __bf16* xbf   = (__bf16*)carve(XSZ * sizeof(__bf16));   // layer-0 input
  __bf16* midbf = (__bf16*)carve(XSZ * sizeof(__bf16));   // layer-0 out / l1 in
  __bf16* gxbuf = (__bf16*)carve(GXSZ * sizeof(__bf16));  // reused per layer
  __bf16* hbf0  = (__bf16*)carve(SSZ * sizeof(__bf16));
  __bf16* hbf1  = (__bf16*)carve(SSZ * sizeof(__bf16));
  float*  hf0   = (float*)carve(SSZ * sizeof(float));
  float*  hf1   = (float*)carve(SSZ * sizeof(float));
  float*  cf0   = (float*)carve(SSZ * sizeof(float));
  float*  cf1   = (float*)carve(SSZ * sizeof(float));

  // 1) pack weights (4 matrices)
  {
    int blocks = (int)((WSZ + 255) / 256);
    for (int l = 0; l < NLAYER; ++l) {
      pack_weight_kernel<<<blocks, 256, 0, stream>>>(Wih + (size_t)l * WSZ,
                                                     WihP + (size_t)l * WSZ,
                                                     HID, GATES);
      pack_weight_kernel<<<blocks, 256, 0, stream>>>(Whh + (size_t)l * WSZ,
                                                     WhhP + (size_t)l * WSZ,
                                                     HID, GATES);
    }
  }

  // 2) x -> bf16
  f32_to_bf16_kernel<<<(int)((XSZ + 255) / 256), 256, 0, stream>>>(x, xbf, XSZ);

  __bf16* hbf[2] = {hbf0, hbf1};
  float*  hf[2]  = {hf0, hf1};
  float*  cf[2]  = {cf0, cf1};

  for (int l = 0; l < NLAYER; ++l) {
    const __bf16* inbf = (l == 0) ? xbf : midbf;

    // 3a) gx = in @ Wih + bias   (M=65536, K=1024, N=4096)
    {
      dim3 grid(GATES / 128, (unsigned)((T_STEPS * BATCH) / 128));
      gx_gemm_kernel<<<grid, 256, 0, stream>>>(inbf, WihP + (size_t)l * WSZ,
                                               bias + (size_t)l * GATES, gxbuf);
    }

    // 3b) zero initial states (parity-0 buffers)
    zero_states_kernel<<<(int)(SSZ / 256), 256, 0, stream>>>(hbf[0], hf[0], cf[0]);

    // 3c) 512 sequential steps (double-buffered states)
    for (int t = 0; t < T_STEPS; ++t) {
      int pi = t & 1;       // read parity
      int po = pi ^ 1;      // write parity
      float*  o32 = (l == NLAYER - 1) ? out + (size_t)t * SSZ : nullptr;
      __bf16* obf = (l == 0) ? midbf + (size_t)t * SSZ : nullptr;
      lstm_step_kernel<<<32, 256, 0, stream>>>(
          gxbuf + (size_t)t * BATCH * GATES, WhhP + (size_t)l * WSZ,
          hbf[pi], hf[pi], cf[pi], lengths, t,
          hbf[po], hf[po], cf[po], o32, obf);
    }

    // 3d) final states (after 512 steps, state is in parity-0 buffers)
    float* h_out = out + (size_t)T_STEPS * SSZ + (size_t)l * SSZ;
    float* c_out = out + (size_t)T_STEPS * SSZ + (size_t)NLAYER * SSZ +
                   (size_t)l * SSZ;
    copy_states_kernel<<<(int)(SSZ / 256), 256, 0, stream>>>(hf[0], cf[0],
                                                             h_out, c_out);
  }
}
